// HEAD_54503134986516
// MI455X (gfx1250) — matrix operand order repaired and checked by reading, not compiled, not run on hardware
//
#include <hip/hip_runtime.h>
#include <hip/hip_bf16.h>

typedef __attribute__((ext_vector_type(16))) _Float16 v16h;
typedef __attribute__((ext_vector_type(8)))  float    v8f;

union V16U { float4 f4[2]; v16h v; };

// ---------------------------------------------------------------- utilities
__global__ void k_zero(float* __restrict__ p, long n) {
  long i = (long)blockIdx.x * blockDim.x + threadIdx.x;
  long s = (long)gridDim.x * blockDim.x;
  for (; i < n; i += s) p[i] = 0.0f;
}

__global__ void k_tohalf(const float* __restrict__ in, _Float16* __restrict__ out, long n) {
  long i = (long)blockIdx.x * blockDim.x + threadIdx.x;
  long s = (long)gridDim.x * blockDim.x;
  for (; i < n; i += s) out[i] = (_Float16)in[i];
}

// ------------------------------------------------ edge conv (WMMA f16 path)
// Augmented K=96 matmul: feat = [x[src](64) | dx | dy | 1 | 0...0]
//                        W    = [Wl[:, :64] | Wl[:,64] | Wl[:,65] | bl | 0...0]
// m[e][o] = relu( (W · feat)[o] ); agg[dst[e]][o] = max(agg, m)
// relu >= 0  ->  int atomicMax on float bits against 0-initialized agg.
// Invalid edges scatter to dump row N. agg stride = MTILES*16 (padded).
// No block barriers in the edge loop: sFeat tiles are wave-private and CDNA5
// LDS processes a wave's DS ops in order, so a compile-time wave_barrier is
// enough for the cross-lane staging; atomics stay fire-and-forget.
template <int MTILES>
__global__ __launch_bounds__(256)
void k_edge_conv(const _Float16* __restrict__ xh,   // N x 64 (f16)
                 const float* __restrict__ pos,     // N x 3
                 const int* __restrict__ eidx,      // 2 x E
                 const float* __restrict__ wl,      // OUT x 66
                 const float* __restrict__ bl,      // OUT
                 float* __restrict__ agg,           // (N+1) x MTILES*16 (zeroed)
                 int E, int G, int iters, int OUT, int Ndump)
{
  constexpr int OUT_PAD = MTILES * 16;
  constexpr bool HOIST_A = (MTILES <= 4);   // keep weight fragments in VGPRs
  __shared__ __align__(16) _Float16 sWl[OUT_PAD * 96];   // augmented weights
  __shared__ __align__(16) _Float16 sFeat[8][16 * 96];   // per-wave edge tiles

  const int tid  = threadIdx.x;
  const int wave = tid >> 5;
  const int lane = tid & 31;

  // augmented weights: K 0..63 = Wl[:, :64]; 64,65 = dpos cols; 66 = bias; 67.. = 0
  for (int i = tid; i < OUT_PAD * 96; i += 256) {
    int o = i / 96, k = i % 96;
    float w = 0.0f;
    if (o < OUT) {
      if (k < 66)       w = wl[o * 66 + k];
      else if (k == 66) w = bl[o];
    }
    sWl[i] = (_Float16)w;
  }
  // constant tail of feature tiles: ch 66 = 1, ch 64,65 set per-iter, rest 0
  for (int i = lane; i < 16 * 32; i += 32) {
    int e = i >> 5, k = 64 + (i & 31);
    sFeat[wave][e * 96 + k] = (_Float16)((k == 66) ? 1.0f : 0.0f);
  }
  __syncthreads();

  const int* __restrict__ srcArr = eidx;
  const int* __restrict__ dstArr = eidx + E;

  const int eb    = lane & 15;             // edge within 16-group (D column)
  const int koffA = (lane >= 16) ? 8 : 0;  // A fragment K layout (ISA 7.12.2)
  const int koffB = (lane >= 16) ? 8 : 0;  // B fragment K layout
  const int mlo   = (lane >= 16) ? 8 : 0;  // D fragment M offset
  const _Float16* fB = &sFeat[wave][eb * 96];

  // hoist loop-invariant weight fragments into registers where they fit
  V16U af[HOIST_A ? MTILES : 1][3];
  if constexpr (HOIST_A) {
    #pragma unroll
    for (int mt = 0; mt < MTILES; ++mt) {
      const _Float16* fA = &sWl[(mt * 16 + (lane & 15)) * 96];
      #pragma unroll
      for (int kc = 0; kc < 3; ++kc) {
        af[mt][kc].f4[0] = *(const float4*)(fA + kc * 32 + koffA);
        af[mt][kc].f4[1] = *(const float4*)(fA + kc * 32 + koffA + 16);
      }
    }
  }

  for (int it = 0; it < iters; ++it) {
    int group  = (it * gridDim.x + blockIdx.x) * 8 + wave;
    long eg    = (long)group * 16 + eb;
    bool gvalid = (group < G) && (eg < E);

    int s = 0, d = Ndump;
    if (gvalid) { s = srcArr[eg]; d = dstArr[eg]; }
    if (lane < 16) {
      float dx = 0.f, dy = 0.f;
      if (gvalid) {
        dx = pos[(long)s * 3 + 0] - pos[(long)d * 3 + 0];
        dy = pos[(long)s * 3 + 1] - pos[(long)d * 3 + 1];
      }
      _Float16* tp = &sFeat[wave][eb * 96 + 64];
      tp[0] = (_Float16)dx;
      tp[1] = (_Float16)dy;
    }
    // gather 64 f16 features: lanes 0-15 lower 32 ch, lanes 16-31 upper 32 ch
    {
      int cbase = (lane >> 4) * 32;
      const float4* rp = (const float4*)(xh + (long)s * 64 + cbase);
      float4* wp = (float4*)(&sFeat[wave][eb * 96 + cbase]);
      #pragma unroll
      for (int j = 0; j < 4; ++j) wp[j] = rp[j];
    }
    __builtin_amdgcn_wave_barrier();   // compile-time order: stores before loads

    // B fragments: one load per K-chunk, reused by every M-tile
    V16U bf[3];
    #pragma unroll
    for (int kc = 0; kc < 3; ++kc) {
      bf[kc].f4[0] = *(const float4*)(fB + kc * 32 + koffB);
      bf[kc].f4[1] = *(const float4*)(fB + kc * 32 + koffB + 16);
    }

    const long rowBase = (long)d * OUT_PAD;
    #pragma unroll
    for (int mt = 0; mt < MTILES; ++mt) {
      v8f acc = {};
      if constexpr (HOIST_A) {
        #pragma unroll
        for (int kc = 0; kc < 3; ++kc)
          acc = __builtin_amdgcn_wmma_f32_16x16x32_f16(
              false, af[mt][kc].v, false, bf[kc].v, (short)0, acc, false, false);
      } else {
        const _Float16* fA = &sWl[(mt * 16 + (lane & 15)) * 96];
        #pragma unroll
        for (int kc = 0; kc < 3; ++kc) {
          V16U ua;
          ua.f4[0] = *(const float4*)(fA + kc * 32 + koffA);
          ua.f4[1] = *(const float4*)(fA + kc * 32 + koffA + 16);
          acc = __builtin_amdgcn_wmma_f32_16x16x32_f16(
              false, ua.v, false, bf[kc].v, (short)0, acc, false, false);
        }
      }
      float* ap = agg + rowBase + mt * 16 + mlo;
      #pragma unroll
      for (int r = 0; r < 8; ++r) {
        float v = fmaxf(acc[r], 0.0f);
        atomicMax((int*)(ap + r), __float_as_int(v));
      }
    }
    __builtin_amdgcn_wave_barrier();   // keep next gather after this iter's loads
  }
}

// ---------------------------------------------- node-level GEMM (OUT x OUT)
__global__ void k_node_linear(const float* __restrict__ agg, const float* __restrict__ wg,
                              const float* __restrict__ bg, float* __restrict__ y,
                              int nrows, int OUT, int aggStride)
{
  extern __shared__ float sW[];
  for (int i = threadIdx.x; i < OUT * OUT; i += blockDim.x) sW[i] = wg[i];
  __syncthreads();
  long total  = (long)nrows * OUT;
  long i      = (long)blockIdx.x * blockDim.x + threadIdx.x;
  long stride = (long)gridDim.x * blockDim.x;
  for (; i < total; i += stride) {
    int n = (int)(i / OUT);
    int o = (int)(i % OUT);
    const float* ar = agg + (long)n * aggStride;
    const float* wr = sW + o * OUT;
    float acc = bg[o];
    for (int k = 0; k < OUT; ++k) acc += ar[k] * wr[k];
    y[i] = acc;
  }
}

// ------------------------------------------------------------- batch norm
__global__ void k_colstats(const float* __restrict__ y, float* __restrict__ sums,
                           float* __restrict__ sumsq, int nrows)
{
  // blockDim and grid*block are multiples of 64 -> each thread owns one column
  int c = threadIdx.x & 63;
  long i      = (long)blockIdx.x * blockDim.x + threadIdx.x;
  long stride = (long)gridDim.x * blockDim.x;
  float s = 0.f, s2 = 0.f;
  for (; i < (long)nrows * 64; i += stride) {
    float v = y[i];
    s += v; s2 += v * v;
  }
  atomicAdd(&sums[c], s);
  atomicAdd(&sumsq[c], s2);
}

__global__ void k_bnrelu(float* __restrict__ y, const float* __restrict__ sums,
                         const float* __restrict__ sumsq, const float* __restrict__ g,
                         const float* __restrict__ b, int nrows)
{
  long i      = (long)blockIdx.x * blockDim.x + threadIdx.x;
  long stride = (long)gridDim.x * blockDim.x;
  float inv_n = 1.0f / (float)nrows;
  for (; i < (long)nrows * 64; i += stride) {
    int c = (int)(i & 63);
    float mu  = sums[c] * inv_n;
    float var = sumsq[c] * inv_n - mu * mu;
    float v = (y[i] - mu) * rsqrtf(var + 1e-5f) * g[c] + b[c];
    y[i] = fmaxf(v, 0.0f);
  }
}

// ----------------------------------------------------------------- driver
extern "C" void kernel_launch(void* const* d_in, const int* in_sizes, int n_in,
                              void* d_out, int out_size, void* d_ws, size_t ws_size,
                              hipStream_t stream)
{
  (void)in_sizes; (void)n_in; (void)out_size; (void)ws_size;
  const int N = 100000, E = 1600000;

  const float* x   = (const float*)d_in[0];
  const float* pos = (const float*)d_in[1];
  const int*   eix = (const int*)d_in[2];
  auto P = [&](int i) { return (const float*)d_in[i]; };

  // workspace layout
  char* ws = (char*)d_ws;
  size_t off = 0;
  auto wsalloc = [&](size_t bytes) { void* p = ws + off; off = (off + bytes + 255) & ~(size_t)255; return p; };
  _Float16* xh    = (_Float16*)wsalloc((size_t)N * 64 * sizeof(_Float16));
  float*    agg   = (float*)wsalloc((size_t)(N + 1) * 112 * sizeof(float));
  float*    hbuf  = (float*)wsalloc((size_t)N * 64 * sizeof(float));
  float*    x1    = (float*)wsalloc((size_t)N * 64 * sizeof(float));
  float*    x2    = (float*)wsalloc((size_t)N * 64 * sizeof(float));
  float*    stats = (float*)wsalloc(256 * sizeof(float)); // sums[0:64], sumsq[64:128]

  float* outCls = (float*)d_out;                 // N x 101
  float* outReg = outCls + (size_t)N * 101;      // N x 4
  float* outObj = outReg + (size_t)N * 4;        // N x 1

  const int TB = 256;
  const int G = (E + 15) / 16;
  const int EBLK = 1024;
  const int iters = (G + EBLK * 8 - 1) / (EBLK * 8);

  auto run_edge = [&](const _Float16* feats, const float* wl, const float* bl,
                      int OUT, int mtiles) {
    long stridep = (long)mtiles * 16;
    k_zero<<<2048, TB, 0, stream>>>(agg, (long)(N + 1) * stridep);
    switch (mtiles) {
      case 1:
        k_edge_conv<1><<<EBLK, TB, 0, stream>>>(feats, pos, eix, wl, bl, agg,
                                                E, G, iters, OUT, N);
        break;
      case 4:
        k_edge_conv<4><<<EBLK, TB, 0, stream>>>(feats, pos, eix, wl, bl, agg,
                                                E, G, iters, OUT, N);
        break;
      default:
        k_edge_conv<7><<<EBLK, TB, 0, stream>>>(feats, pos, eix, wl, bl, agg,
                                                E, G, iters, OUT, N);
        break;
    }
  };
  auto run_node = [&](const float* wg, const float* bg, float* y, int OUT, int mtiles) {
    k_node_linear<<<2048, TB, (size_t)OUT * OUT * sizeof(float), stream>>>(
        agg, wg, bg, y, N, OUT, mtiles * 16);
  };
  auto run_bn = [&](float* y, const float* g, const float* b) {
    k_zero<<<1, 128, 0, stream>>>(stats, 128);
    k_colstats<<<1024, TB, 0, stream>>>(y, stats, stats + 64, N);
    k_bnrelu<<<2048, TB, 0, stream>>>(y, stats, stats + 64, g, b, N);
  };

  // stem: h = bn_relu(conv(x))
  k_tohalf<<<2048, TB, 0, stream>>>(x, xh, (long)N * 64);
  run_edge(xh, P(3), P(4), 64, 4);
  run_node(P(5), P(6), hbuf, 64, 4);
  run_bn(hbuf, P(7), P(8));

  // c1, c2 share f16(h)
  k_tohalf<<<2048, TB, 0, stream>>>(hbuf, xh, (long)N * 64);
  run_edge(xh, P(9), P(10), 64, 4);
  run_node(P(11), P(12), x1, 64, 4);
  run_bn(x1, P(13), P(14));
  run_edge(xh, P(15), P(16), 64, 4);
  run_node(P(17), P(18), x2, 64, 4);
  run_bn(x2, P(19), P(20));

  // reg & obj from x1
  k_tohalf<<<2048, TB, 0, stream>>>(x1, xh, (long)N * 64);
  run_edge(xh, P(21), P(22), 4, 1);
  run_node(P(23), P(24), outReg, 4, 1);
  run_edge(xh, P(29), P(30), 1, 1);
  run_node(P(31), P(32), outObj, 1, 1);

  // cls from x2
  k_tohalf<<<2048, TB, 0, stream>>>(x2, xh, (long)N * 64);
  run_edge(xh, P(25), P(26), 101, 7);
  run_node(P(27), P(28), outCls, 101, 7);
}
